// DecoderLayer_42880953484173
// MI455X (gfx1250) — matrix operand order repaired and checked by reading, not compiled, not run on hardware
//
#include <hip/hip_runtime.h>
#include <hip/hip_bf16.h>
#include <math.h>

// ---------------------------------------------------------------------------
// Types for CDNA5 WMMA (wave32, 16x16x32 bf16 -> f32)
// ---------------------------------------------------------------------------
typedef __bf16 bf16_t;
typedef __attribute__((ext_vector_type(16))) __bf16 v16bf;
typedef __attribute__((ext_vector_type(8)))  __bf16 v8bf;
typedef __attribute__((ext_vector_type(8)))  float  v8f;
typedef __attribute__((ext_vector_type(4)))  unsigned int u32x4;
typedef __attribute__((ext_vector_type(4)))  int i32x4;
typedef __attribute__((ext_vector_type(8)))  int i32x8;

#define D_MODEL 1024
#define NHEAD   16
#define DHEAD   64
#define DFF_    4096
#define TT      2048
#define BB      2
#define MROWS   (BB * TT)   // 4096

// Tensor Data Mover available? (device pass only; host pass parses fallback)
#if defined(__AMDGCN__) && __has_builtin(__builtin_amdgcn_tensor_load_to_lds)
#define USE_TDM 1
#else
#define USE_TDM 0
#endif

static __device__ __forceinline__ v8f wmma_bf16(v16bf a, v16bf b, v8f c) {
  // D = A(16x32) * B(32x16) + C(16x16 f32)
  return __builtin_amdgcn_wmma_f32_16x16x32_bf16(
      /*neg_a=*/false, a, /*neg_b=*/false, b,
      /*c_mod=*/(short)0, c, /*reuse_a=*/false, /*reuse_b=*/false);
}

static __device__ __forceinline__ v8f vzero8() {
  v8f z = {0.f, 0.f, 0.f, 0.f, 0.f, 0.f, 0.f, 0.f};
  return z;
}

// ---------------------------------------------------------------------------
// fp32 -> bf16 elementwise convert (for weights)
// ---------------------------------------------------------------------------
__global__ void f32_to_bf16_kernel(const float* __restrict__ in,
                                   bf16_t* __restrict__ out, int n) {
  int i = blockIdx.x * blockDim.x + threadIdx.x;
  if (i < n) out[i] = (bf16_t)in[i];
}

// ---------------------------------------------------------------------------
// LayerNorm over D=1024, one 256-thread block per row, bf16 output
// ---------------------------------------------------------------------------
__global__ __launch_bounds__(256)
void layernorm_bf16_kernel(const float* __restrict__ x,
                           const float* __restrict__ w,
                           const float* __restrict__ b,
                           bf16_t* __restrict__ out) {
  __shared__ float red[256];
  const int tid = threadIdx.x;
  const int row = blockIdx.x;
  const float* xr = x + (size_t)row * D_MODEL;

  float4 xv = *(const float4*)(xr + tid * 4);
  float s = xv.x + xv.y + xv.z + xv.w;
  red[tid] = s;
  __syncthreads();
  for (int off = 128; off > 0; off >>= 1) {
    if (tid < off) red[tid] += red[tid + off];
    __syncthreads();
  }
  float mu = red[0] * (1.0f / D_MODEL);
  __syncthreads();

  float d0 = xv.x - mu, d1 = xv.y - mu, d2 = xv.z - mu, d3 = xv.w - mu;
  red[tid] = d0 * d0 + d1 * d1 + d2 * d2 + d3 * d3;
  __syncthreads();
  for (int off = 128; off > 0; off >>= 1) {
    if (tid < off) red[tid] += red[tid + off];
    __syncthreads();
  }
  float rstd = rsqrtf(red[0] * (1.0f / D_MODEL) + 1e-5f);

  int c = tid * 4;
  bf16_t* orow = out + (size_t)row * D_MODEL;
  orow[c + 0] = (bf16_t)(d0 * rstd * w[c + 0] + b[c + 0]);
  orow[c + 1] = (bf16_t)(d1 * rstd * w[c + 1] + b[c + 1]);
  orow[c + 2] = (bf16_t)(d2 * rstd * w[c + 2] + b[c + 2]);
  orow[c + 3] = (bf16_t)(d3 * rstd * w[c + 3] + b[c + 3]);
}

// ---------------------------------------------------------------------------
// WMMA GEMM: C[M,N] = A[M,K](bf16) * W[N,K]^T(bf16) + bias, fused epilogue.
// Block: 256 threads (8 waves). Block tile 64(M) x 256(N).
// Waves arranged 2(M) x 4(N); each wave computes 32x64 via 2x4 WMMA frags
// (8 WMMAs per 32-deep K-step against 4 global B-frags + 4 LDS A-loads).
// A tile (64x32) staged in LDS each K-step by the Tensor Data Mover (wave 0
// issues tensor_load_to_lds with a 2-D D#, waits TENSORcnt, barrier
// publishes); falls back to a cooperative vector copy when TDM builtin is
// unavailable. B fragments stream from global (24MB bf16 weights << 192MB L2)
// with global_prefetch for the next K-step.
// flags: 1=gelu(exact erf) 2=add residual 4=write f32 8=write bf16
// ---------------------------------------------------------------------------
__global__ __launch_bounds__(256)
void gemm_wmma_kernel(const bf16_t* __restrict__ A, const bf16_t* __restrict__ W,
                      const float* __restrict__ bias, const float* __restrict__ resid,
                      float* __restrict__ Cf, bf16_t* __restrict__ Cbf,
                      int M, int N, int K, int flags) {
  __shared__ bf16_t As[64 * 32];

  const int tid  = threadIdx.x;
  const int lane = tid & 31;
  const int wave = tid >> 5;
  const int wm   = wave >> 2;    // 0..1
  const int wn   = wave & 3;     // 0..3
  const int lh   = lane >> 4;    // half of wave: 0/1
  const int l16  = lane & 15;

  const int blockM = blockIdx.y * 64;
  const int blockN = blockIdx.x * 256;

  const int kbaseA = lh * 16;    // A frag: 16 contiguous K per half
  const int kbaseB = lh * 16;    // B frag: 16 contiguous K per half

  v8f acc[2][4];
  for (int mi = 0; mi < 2; mi++)
    for (int ni = 0; ni < 4; ni++) acc[mi][ni] = vzero8();

#if USE_TDM
  // ---- Tensor DMA descriptor (2-D tile: 32 K-elems x 64 rows, bf16) ----
  // Group 0: count=1 | lds_addr | global_addr(57b) | type=2
  // Group 1: data_size=2B, tensor_dim0=K, tensor_dim1=rows, tile 32x64,
  //          tensor_dim0_stride=K  (CDNA5 ISA)
  const unsigned ldsOff = (unsigned)(size_t)(void*)As;
  const unsigned long long gbase =
      (unsigned long long)(size_t)(A + (size_t)blockM * K);
  u32x4 g0;
  g0[0] = 1u;                       // count=1, user descriptor
  g0[1] = ldsOff;                   // LDS destination
  g0[2] = 0u; g0[3] = 0u;           // global addr filled per K-step
  i32x8 g1;
  const unsigned rowsLeft = (unsigned)(M - blockM);
  g1[0] = (int)(1u << 16);                                   // data_size=1 (2B)
  g1[1] = (int)(((unsigned)K & 0xFFFFu) << 16);              // dim0[15:0]
  g1[2] = (int)(((unsigned)K >> 16) | ((rowsLeft & 0xFFFFu) << 16));
  g1[3] = (int)((rowsLeft >> 16) | (32u << 16));             // tile_dim0=32
  g1[4] = 64;                                                // tile_dim1=64
  g1[5] = (int)(unsigned)K;                                  // dim0_stride lo
  g1[6] = 0;
  g1[7] = 0;
  const i32x4 z4 = {0, 0, 0, 0};
#if __clang_major__ >= 23
  const i32x8 z8 = {0, 0, 0, 0, 0, 0, 0, 0};
#endif
#else
  // cooperative A staging: 64 rows x 32 k; thread -> (row = tid>>2, 8 cols)
  const int ldsRow = tid >> 2;
  const int ldsCol = (tid & 3) * 8;
  const bf16_t* Aload = A + (size_t)(blockM + ldsRow) * K + ldsCol;
#endif

  for (int k0 = 0; k0 < K; k0 += 32) {
    __syncthreads();   // previous tile fully consumed
#if USE_TDM
    if (wave == 0) {
      unsigned long long ga = gbase + (unsigned long long)k0 * 2ull;
      g0[2] = (unsigned)ga;
      g0[3] = (unsigned)((ga >> 32) & 0x01FFFFFFull) | (2u << 30);
#if __clang_major__ >= 23
      __builtin_amdgcn_tensor_load_to_lds(g0, g1, z4, z4, z8, 0);
#else
      __builtin_amdgcn_tensor_load_to_lds(g0, g1, z4, z4, 0);
#endif
      __builtin_amdgcn_s_wait_tensorcnt(0);
    }
#else
    *(v8bf*)(As + ldsRow * 32 + ldsCol) = *(const v8bf*)(Aload + k0);
#endif
    __syncthreads();   // tile visible to all 8 waves

    v16bf bfrag[4];
    for (int ni = 0; ni < 4; ni++) {
      int n = blockN + wn * 64 + ni * 16 + l16;
      const bf16_t* wrow = W + (size_t)n * K + kbaseB;
      bfrag[ni] = *(const v16bf*)(wrow + k0);
      if (k0 + 32 < K)  // prefetch next K-step's B line into L2/L0
        __builtin_prefetch((const void*)(wrow + k0 + 32), 0, 3);
    }
    for (int mi = 0; mi < 2; mi++) {
      union { v16bf v; v8bf h2[2]; } a;
      const bf16_t* ap = As + (wm * 32 + mi * 16 + l16) * 32 + kbaseA;
      a.h2[0] = *(const v8bf*)(ap);
      a.h2[1] = *(const v8bf*)(ap + 8);
      for (int ni = 0; ni < 4; ni++)
        acc[mi][ni] = wmma_bf16(a.v, bfrag[ni], acc[mi][ni]);
    }
  }

  const bool do_gelu = flags & 1;
  const bool do_res  = flags & 2;
  const bool wr_f32  = flags & 4;
  const bool wr_bf16 = flags & 8;

  for (int mi = 0; mi < 2; mi++) {
    for (int ni = 0; ni < 4; ni++) {
      int col = blockN + wn * 64 + ni * 16 + l16;
      float bcol = bias ? bias[col] : 0.0f;
      for (int r = 0; r < 8; r++) {
        int row = blockM + wm * 32 + mi * 16 + lh * 8 + r;
        float v = acc[mi][ni][r] + bcol;
        if (do_gelu) v = 0.5f * v * (1.0f + erff(v * 0.70710678118654752f));
        size_t idx = (size_t)row * N + col;
        if (do_res) v += resid[idx];
        if (wr_f32)  Cf[idx]  = v;
        if (wr_bf16) Cbf[idx] = (bf16_t)v;
      }
    }
  }
}

// ---------------------------------------------------------------------------
// Causal flash attention, WMMA for S=QK^T and O=PV.
// Grid: (T/64, B*H). Block: 128 threads (4 waves); each wave owns 16 queries.
// qkv layout: [B*T, 3*D]; head slice h*64 within each D-third.
// Online softmax stats kept per-lane in C-frag row mapping (row=r+8*half),
// reduced across the 16 column lanes with __shfl_xor (wave32 halves).
// P is rewritten through wave-private LDS into A-fragment layout; V tile is
// transposed through LDS into B-fragment layout.
// ---------------------------------------------------------------------------
__global__ __launch_bounds__(128)
void attn_flash_wmma_kernel(const bf16_t* __restrict__ qkv,
                            bf16_t* __restrict__ y) {
  __shared__ bf16_t Pb[4][16 * 32];
  __shared__ bf16_t Vt[4][64 * 32];

  const int lane = threadIdx.x & 31;
  const int wave = threadIdx.x >> 5;
  const int lh   = lane >> 4;
  const int l16  = lane & 15;

  const int b = blockIdx.y / NHEAD;
  const int h = blockIdx.y % NHEAD;
  const int q0 = blockIdx.x * 64 + wave * 16;

  const int LD = 3 * D_MODEL;
  const size_t base = ((size_t)b * TT) * LD + h * DHEAD;
  const bf16_t* Q  = qkv + base;
  const bf16_t* Kp = qkv + base + D_MODEL;
  const bf16_t* Vp = qkv + base + 2 * D_MODEL;

  const int kbaseA = lh * 16;
  const int kbaseB = lh * 16;

  // Q A-fragments for dh blocks {0..31, 32..63}
  union { v16bf v; v8bf h2[2]; } qa[2];
  {
    const bf16_t* qrow = Q + (size_t)(q0 + l16) * LD;
    for (int dk = 0; dk < 2; dk++) {
      qa[dk].h2[0] = *(const v8bf*)(qrow + dk * 32 + kbaseA);
      qa[dk].h2[1] = *(const v8bf*)(qrow + dk * 32 + kbaseA + 8);
    }
  }

  v8f o[4];
  for (int nf = 0; nf < 4; nf++) o[nf] = vzero8();
  float mrun[8], lrun[8];
  for (int r = 0; r < 8; r++) { mrun[r] = -1e30f; lrun[r] = 0.0f; }

  bf16_t* Pw = Pb[wave];
  bf16_t* Vw = Vt[wave];

  const int njt = (q0 + 15 + 32) / 32;  // 32-key tiles covering keys <= q0+15
  for (int j = 0; j < njt; j++) {
    const int key0 = j * 32;

    // stage V^T: Vw[dh*32 + kk] = V[key0+kk][dh]   (one key per lane)
    {
      const bf16_t* vrow = Vp + (size_t)(key0 + lane) * LD;
      for (int d = 0; d < DHEAD; d++) Vw[d * 32 + lane] = vrow[d];
    }

    // S = Q * K^T  (two 16-key C-frags, K-dim = 64 in two WMMA steps)
    v8f s[2];
    for (int kf = 0; kf < 2; kf++) {
      s[kf] = vzero8();
      const int key = key0 + kf * 16 + l16;
      const bf16_t* krow = Kp + (size_t)key * LD;
      for (int dk = 0; dk < 2; dk++) {
        v16bf kb = *(const v16bf*)(krow + dk * 32 + kbaseB);
        s[kf] = wmma_bf16(qa[dk].v, kb, s[kf]);
      }
    }

    // scale, causal mask, online softmax, stash P (bf16) into LDS
    for (int r = 0; r < 8; r++) {
      const int qrow = q0 + lh * 8 + r;
      float v0 = s[0][r] * 0.125f;          // 1/sqrt(64)
      float v1 = s[1][r] * 0.125f;
      if (key0 + l16 > qrow)      v0 = -1e30f;
      if (key0 + 16 + l16 > qrow) v1 = -1e30f;
      float mx = fmaxf(v0, v1);
      for (int off = 8; off > 0; off >>= 1)
        mx = fmaxf(mx, __shfl_xor(mx, off, 32));
      const float mnew = fmaxf(mrun[r], mx);
      const float p0 = __expf(v0 - mnew);
      const float p1 = __expf(v1 - mnew);
      const float alpha = __expf(mrun[r] - mnew);
      float rsum = p0 + p1;
      for (int off = 8; off > 0; off >>= 1)
        rsum += __shfl_xor(rsum, off, 32);
      lrun[r] = lrun[r] * alpha + rsum;
      mrun[r] = mnew;
      for (int nf = 0; nf < 4; nf++) o[nf][r] *= alpha;
      Pw[(lh * 8 + r) * 32 + l16]      = (bf16_t)p0;
      Pw[(lh * 8 + r) * 32 + 16 + l16] = (bf16_t)p1;
    }

    // O += P(16x32) * V(32x64): P from LDS as A-frag, V^T from LDS as B-frags
    union { v16bf v; v8bf h2[2]; } pa;
    const bf16_t* pp = Pw + l16 * 32 + kbaseA;
    pa.h2[0] = *(const v8bf*)(pp);
    pa.h2[1] = *(const v8bf*)(pp + 8);
    for (int nf = 0; nf < 4; nf++) {
      v16bf vb = *(const v16bf*)(Vw + (nf * 16 + l16) * 32 + kbaseB);
      o[nf] = wmma_bf16(pa.v, vb, o[nf]);
    }
  }

  // normalize and emit y (heads re-interleaved into [B*T, D])
  for (int nf = 0; nf < 4; nf++) {
    for (int r = 0; r < 8; r++) {
      const int qrow = q0 + lh * 8 + r;
      const float val = o[nf][r] / lrun[r];
      y[((size_t)(b * TT + qrow)) * D_MODEL + h * DHEAD + nf * 16 + l16] =
          (bf16_t)val;
    }
  }
}

// ---------------------------------------------------------------------------
// Host orchestration
// ---------------------------------------------------------------------------
extern "C" void kernel_launch(void* const* d_in, const int* in_sizes, int n_in,
                              void* d_out, int out_size, void* d_ws, size_t ws_size,
                              hipStream_t stream) {
  (void)in_sizes; (void)n_in; (void)out_size; (void)ws_size;

  const float* x      = (const float*)d_in[0];
  const float* ln1_w  = (const float*)d_in[1];
  const float* ln1_b  = (const float*)d_in[2];
  const float* qkv_w  = (const float*)d_in[3];
  const float* qkv_b  = (const float*)d_in[4];
  const float* out_w  = (const float*)d_in[5];
  const float* out_b  = (const float*)d_in[6];
  const float* ln2_w  = (const float*)d_in[7];
  const float* ln2_b  = (const float*)d_in[8];
  const float* mlp_w1 = (const float*)d_in[9];
  const float* mlp_b1 = (const float*)d_in[10];
  const float* mlp_w2 = (const float*)d_in[11];
  const float* mlp_b2 = (const float*)d_in[12];

  char* ws = (char*)d_ws;
  size_t off = 0;
  auto alloc = [&](size_t bytes) -> void* {
    void* p = ws + off;
    off += (bytes + 255) & ~(size_t)255;
    return p;
  };

  const int M = MROWS;  // 4096 token rows
  bf16_t* xln  = (bf16_t*)alloc((size_t)M * D_MODEL * 2);
  bf16_t* qkvb = (bf16_t*)alloc((size_t)M * 3 * D_MODEL * 2);
  bf16_t* yb   = (bf16_t*)alloc((size_t)M * D_MODEL * 2);
  float*  x1   = (float*)alloc((size_t)M * D_MODEL * 4);
  bf16_t* hln  = (bf16_t*)alloc((size_t)M * D_MODEL * 2);
  bf16_t* h1   = (bf16_t*)alloc((size_t)M * DFF_ * 2);
  bf16_t* wq   = (bf16_t*)alloc((size_t)3 * D_MODEL * D_MODEL * 2);
  bf16_t* wo   = (bf16_t*)alloc((size_t)D_MODEL * D_MODEL * 2);
  bf16_t* w1   = (bf16_t*)alloc((size_t)DFF_ * D_MODEL * 2);
  bf16_t* w2   = (bf16_t*)alloc((size_t)D_MODEL * DFF_ * 2);

  auto cvt = [&](const float* src, bf16_t* dst, int n) {
    f32_to_bf16_kernel<<<(n + 255) / 256, 256, 0, stream>>>(src, dst, n);
  };
  cvt(qkv_w,  wq, 3 * D_MODEL * D_MODEL);
  cvt(out_w,  wo, D_MODEL * D_MODEL);
  cvt(mlp_w1, w1, DFF_ * D_MODEL);
  cvt(mlp_w2, w2, D_MODEL * DFF_);

  // LN1 -> bf16
  layernorm_bf16_kernel<<<M, 256, 0, stream>>>(x, ln1_w, ln1_b, xln);

  // QKV projection: [M,1024] x [3072,1024]^T  (+bias, bf16 out)
  {
    dim3 g(3 * D_MODEL / 256, M / 64);
    gemm_wmma_kernel<<<g, 256, 0, stream>>>(xln, wq, qkv_b, nullptr,
                                            nullptr, qkvb,
                                            M, 3 * D_MODEL, D_MODEL, /*bf16*/8);
  }

  // Causal flash attention -> y (bf16)
  {
    dim3 g(TT / 64, BB * NHEAD);
    attn_flash_wmma_kernel<<<g, 128, 0, stream>>>(qkvb, yb);
  }

  // Output projection + residual: x1 = x + y*out_w^T + out_b  (f32 out)
  {
    dim3 g(D_MODEL / 256, M / 64);
    gemm_wmma_kernel<<<g, 256, 0, stream>>>(yb, wo, out_b, x,
                                            x1, nullptr,
                                            M, D_MODEL, D_MODEL, /*res|f32*/2 | 4);
  }

  // LN2 -> bf16
  layernorm_bf16_kernel<<<M, 256, 0, stream>>>(x1, ln2_w, ln2_b, hln);

  // MLP up: [M,1024] x [4096,1024]^T, exact-erf GELU, bf16 out
  {
    dim3 g(DFF_ / 256, M / 64);
    gemm_wmma_kernel<<<g, 256, 0, stream>>>(hln, w1, mlp_b1, nullptr,
                                            nullptr, h1,
                                            M, DFF_, D_MODEL, /*gelu|bf16*/1 | 8);
  }

  // MLP down + residual: out = x1 + h1*mlp_w2^T + mlp_b2  (f32 out)
  {
    dim3 g(D_MODEL / 256, M / 64);
    gemm_wmma_kernel<<<g, 256, 0, stream>>>(h1, w2, mlp_b2, x1,
                                            (float*)d_out, nullptr,
                                            M, D_MODEL, DFF_, /*res|f32*/2 | 4);
  }
}
